// SubInfoNCEloss_37228776521950
// MI455X (gfx1250) — matrix operand order repaired and checked by reading, not compiled, not run on hardware
//
#include <hip/hip_runtime.h>

// ---------------------------------------------------------------------------
// Skip-gram negative-sampling loss (word2vec SGNS) for MI455X / gfx1250.
//
// Roofline: 0.23 GFLOP vs ~461 MB of gathered embedding rows -> pure
// bandwidth problem (tables are L2-resident: 2 x 51.2 MB < 192 MB L2).
// WMMA f32_16x16x32_f16 is used as a zero-shuffle dot-product engine:
// B = i_vec replicated across all 16 columns, so every column of D is the
// row-dot and the matrix unit performs the cross-lane reduction for free.
// log_sigmoid uses raw v_exp_f32/v_log_f32 (base-2) -- args are range-safe
// by construction, so no libm range-reduction/denorm fixup code is needed.
// ---------------------------------------------------------------------------

#define VOCAB  100000
#define DIM    128
#define BATCH  4096
#define CTX    20
#define NNEG   10
#define ROWS   (CTX + CTX * NNEG)   // 220 gathered rows per batch element
#define TILES  14                    // ceil(220 / 16), 4 zero-weight pad rows
#define WAVES_PER_BLOCK 8            // 256 threads = 8 wave32

typedef __attribute__((ext_vector_type(16))) _Float16 v16h;
typedef __attribute__((ext_vector_type(8)))  float    v8f;
typedef __attribute__((ext_vector_type(4)))  float    v4f;

__device__ __forceinline__ float log_sigmoid(float x) {
  // log(sigmoid(x)) = min(x,0) - ln(1 + exp(-|x|))
  // exp(-|x|) = exp2(-|x| * log2(e));  arg <= 0  -> bare v_exp_f32 is exact
  // ln(1+e)   = log2(1+e) * ln(2);     1+e in [1,2] -> bare v_log_f32 is safe
  const float LOG2E = 1.44269504088896340736f;
  const float LN2   = 0.69314718055994530942f;
  float e = __builtin_amdgcn_exp2f(-fabsf(x) * LOG2E);
  float l = __builtin_amdgcn_logf(1.0f + e);
  return fminf(x, 0.0f) - l * LN2;
}

__global__ void __launch_bounds__(256)
sgns_wmma_kernel(const int*   __restrict__ i_words,   // [1, B]
                 const int*   __restrict__ o_words,   // [C, B]
                 const int*   __restrict__ n_words,   // [B, C*NEG]
                 const float* __restrict__ W_i,       // [VOCAB, DIM]
                 const float* __restrict__ W_os,      // [VOCAB, DIM]
                 float*       __restrict__ block_partials)
{
  const int lane = threadIdx.x & 31;
  const int wave = threadIdx.x >> 5;
  const int b    = blockIdx.x * WAVES_PER_BLOCK + wave;  // one batch elem / wave
  const int half = lane >> 4;        // 0: lanes 0-15, 1: lanes 16-31
  const int mrow = lane & 15;        // A-matrix row index owned by this lane

  // ---- Build B operand once per wave: i_vec (f16), identical in all 16
  // ---- columns.  B layout (16-bit, 32x16): lanes 0-15 hold K = {0..7, 16..23};
  // ---- lanes 16-31 hold K = {8..15, 24..31}.  Replicated columns => D[m][n] = dot_m.
  const int center = i_words[b];
  const float* iv  = W_i + (size_t)center * DIM;
  v16h Bop[4];
  #pragma unroll
  for (int t = 0; t < 4; ++t) {
    const float* p = iv + 32 * t + half * 8;       // 32B-aligned
    v4f f0 = *(const v4f*)(p +  0);
    v4f f1 = *(const v4f*)(p +  4);
    v4f f2 = *(const v4f*)(p + 16);
    v4f f3 = *(const v4f*)(p + 20);
    #pragma unroll
    for (int j = 0; j < 4; ++j) {
      Bop[t][j +  0] = (_Float16)f0[j];
      Bop[t][j +  4] = (_Float16)f1[j];
      Bop[t][j +  8] = (_Float16)f2[j];
      Bop[t][j + 12] = (_Float16)f3[j];
    }
  }

  float acc = 0.0f;

  for (int tile = 0; tile < TILES; ++tile) {
    // Row gathered by this lane (lanes l and l+16 serve the same A row).
    const int r = tile * 16 + mrow;
    int idx = 0;                                    // pad rows -> zero row 0
    if (r < CTX)        idx = o_words[r * BATCH + b];
    else if (r < ROWS)  idx = n_words[(size_t)b * (CTX * NNEG) + (r - CTX)];
    const float* row = W_os + (size_t)idx * DIM;

    v8f d = {};
    #pragma unroll
    for (int t = 0; t < 4; ++t) {
      // A layout (16-bit, 16x32): lanes 0-15 hold K = {0..7, 16..23} of the
      // 32-wide chunk; lanes 16-31 hold K = {8..15, 24..31}.
      const float* p = row + 32 * t + half * 8;     // 32B-aligned
      v4f a0 = *(const v4f*)(p +  0);
      v4f a1 = *(const v4f*)(p +  4);
      v4f a2 = *(const v4f*)(p + 16);
      v4f a3 = *(const v4f*)(p + 20);
      v16h A;
      #pragma unroll
      for (int j = 0; j < 4; ++j) {
        A[j +  0] = (_Float16)a0[j];
        A[j +  4] = (_Float16)a1[j];
        A[j +  8] = (_Float16)a2[j];
        A[j + 12] = (_Float16)a3[j];
      }
      d = __builtin_amdgcn_wmma_f32_16x16x32_f16(
              /*neg_a=*/false, A, /*neg_b=*/false, Bop[t],
              /*c_mod=*/(short)0, d, /*reuse_a=*/false, /*reuse_b=*/false);
    }

    // D layout: VGPR j, lanes 0-15 -> M=j ; lanes 16-31 -> M=j+8.
    // All 16 columns are identical, so each lane-half holds 16 copies.
    #pragma unroll
    for (int j = 0; j < 8; ++j) {
      const int rr = tile * 16 + half * 8 + j;
      if (rr < ROWS) {
        const float dot = d[j];
        const float w   = (rr < CTX) ? (1.0f / CTX) : (1.0f / NNEG);
        const float s   = (rr < CTX) ? dot : -dot;   // negatives use -W_os row
        acc += w * log_sigmoid(s);
      }
    }
  }

  // Wave sum is 16x the true per-b sum (column replication) -> scale by 1/16.
  #pragma unroll
  for (int off = 16; off > 0; off >>= 1)
    acc += __shfl_xor(acc, off, 32);
  acc *= (1.0f / 16.0f);

  __shared__ float ws[WAVES_PER_BLOCK];
  if (lane == 0) ws[wave] = acc;
  __syncthreads();
  if (threadIdx.x == 0) {
    float s = 0.0f;
    #pragma unroll
    for (int i = 0; i < WAVES_PER_BLOCK; ++i) s += ws[i];
    block_partials[blockIdx.x] = s;   // unconditional write: no pre-zero needed
  }
}

// Deterministic fixed-order final reduction (no float atomics -> bitwise
// reproducible across graph replays).  Writes  -(total)  per the reference.
__global__ void __launch_bounds__(256)
sgns_reduce_kernel(const float* __restrict__ parts, int n,
                   float* __restrict__ out)
{
  __shared__ float sm[256];
  float s = 0.0f;
  for (int i = threadIdx.x; i < n; i += 256) s += parts[i];
  sm[threadIdx.x] = s;
  __syncthreads();
  for (int stride = 128; stride > 0; stride >>= 1) {
    if ((int)threadIdx.x < stride) sm[threadIdx.x] += sm[threadIdx.x + stride];
    __syncthreads();
  }
  if (threadIdx.x == 0) out[0] = -sm[0];
}

extern "C" void kernel_launch(void* const* d_in, const int* in_sizes, int n_in,
                              void* d_out, int out_size, void* d_ws, size_t ws_size,
                              hipStream_t stream) {
  const int*   i_words = (const int*)  d_in[0];
  const int*   o_words = (const int*)  d_in[1];
  const int*   n_words = (const int*)  d_in[2];
  const float* W_i     = (const float*)d_in[3];
  const float* W_os    = (const float*)d_in[4];

  float* partials = (float*)d_ws;               // BATCH/WAVES_PER_BLOCK floats
  const int nblocks = BATCH / WAVES_PER_BLOCK;  // 512

  sgns_wmma_kernel<<<nblocks, 256, 0, stream>>>(
      i_words, o_words, n_words, W_i, W_os, partials);
  sgns_reduce_kernel<<<1, 256, 0, stream>>>(partials, nblocks, (float*)d_out);
}
